// MultiheadAttention_34016140984658
// MI455X (gfx1250) — matrix operand order repaired and checked by reading, not compiled, not run on hardware
//
#include <hip/hip_runtime.h>

// ---------------- problem constants (from reference) ----------------
#define E_   1024
#define H_   16
#define HD_  64
#define DK_  32
#define DC_  128
#define DCP_ 128
#define T_   2048
#define B_   2
#define S_   2048
#define MROWS (T_ * B_)          // 4096 rows for all big GEMMs

typedef __bf16 bf16;
typedef __attribute__((ext_vector_type(8)))  __bf16 v8bf;
typedef __attribute__((ext_vector_type(16))) __bf16 v16bf;
typedef __attribute__((ext_vector_type(8)))  float  v8f;

// ---------------- WMMA helpers (gfx1250 wave32) ----------------
// D = A(16x32 bf16) * B(32x16 bf16) + C(16x16 f32)
static __device__ __forceinline__ v8f wmma_bf16(v16bf a, v16bf b, v8f c) {
  return __builtin_amdgcn_wmma_f32_16x16x32_bf16(
      /*neg_a=*/false, a, /*neg_b=*/false, b,
      /*c_mod=*/(short)0, c, /*reuse_a=*/false, /*reuse_b=*/false);
}

// A-operand 16x32 tile from row-major [*, ld] bf16 matrix.
// lane<16: M=lane, K = {0..7, 16..23}; lane>=16: M=lane-16, K = {8..15, 24..31}
static __device__ __forceinline__ v16bf load_a_tile(const bf16* base, int ld, int lane) {
  const int r  = lane & 15;
  const int h8 = (lane >> 4) << 3;           // 0 or 8
  const v8bf lo = *(const v8bf*)(base + r * ld + h8);
  const v8bf hi = *(const v8bf*)(base + r * ld + h8 + 16);
  v16bf t;
#pragma unroll
  for (int i = 0; i < 8; ++i) { t[i] = lo[i]; t[i + 8] = hi[i]; }
  return t;
}

// B-operand 32x16 tile; source stored row-major [N, K] (each row = one output column).
// lane<16: N=lane, K = {0..7, 16..23}; lane>=16: N=lane-16, K = {8..15, 24..31}
static __device__ __forceinline__ v16bf load_b_tile(const bf16* base, int ld, int lane) {
  const int r   = lane & 15;
  const int h16 = (lane >> 4) << 3;          // 0 or 8
  const v8bf lo = *(const v8bf*)(base + r * ld + h16);
  const v8bf hi = *(const v8bf*)(base + r * ld + h16 + 16);
  v16bf t;
#pragma unroll
  for (int i = 0; i < 8; ++i) { t[i] = lo[i]; t[i + 8] = hi[i]; }
  return t;
}

// ---------------- small prep kernels ----------------
__global__ void f32_to_bf16_k(const float* __restrict__ in, bf16* __restrict__ out, int n) {
  int i = blockIdx.x * blockDim.x + threadIdx.x;
  const int stride = gridDim.x * blockDim.x;
  for (; i < n; i += stride) out[i] = (bf16)in[i];
}

// out[r] = scale * (badd[r] + dot(W[r, 0..K-1], bvec))
__global__ void fold_bias_k(const float* __restrict__ W, const float* __restrict__ bvec,
                            const float* __restrict__ badd, int rows, int K, float scale,
                            float* __restrict__ out) {
  const int r = blockIdx.x * blockDim.x + threadIdx.x;
  if (r >= rows) return;
  float acc = badd[r];
  for (int i = 0; i < K; ++i) acc += W[r * K + i] * bvec[i];
  out[r] = acc * scale;
}

// C[r, j] = scale * sum_i A[r, i] * Bm[i, j]  (small weight-folding GEMMs, fp32)
// grid = (rowsA, N/256); A[r,i] is block-uniform -> scalar broadcast, Bm reads coalesced.
__global__ __launch_bounds__(256) void combine_w_k(const float* __restrict__ A,
                                                   const float* __restrict__ Bm,
                                                   int K, int N, float scale,
                                                   float* __restrict__ outF,
                                                   bf16* __restrict__ outB) {
  const int r = blockIdx.x;
  const int j = blockIdx.y * 256 + threadIdx.x;
  float acc = 0.f;
  for (int i = 0; i < K; ++i) acc += A[r * K + i] * Bm[i * N + j];
  acc *= scale;
  if (outF) outF[r * N + j] = acc;
  else      outB[r * N + j] = (bf16)acc;
}

// ---------------- main bf16 WMMA GEMM ----------------
// C[M=4096, N] = A[M, 1024] * W[N, 1024]^T + bias[N]; per wave: 32x64 output tile.
// Ping-pong double buffering (k-loop body holds both phases; unroll 1 keeps register
// pressure at one live tile set per phase so accumulators stay pinned in place).
// mode 0: store bf16 as QH/KH layout [B,H,T,headDim]   (row=(t*B+b), col=h*headDim+k)
// mode 1: store bf16 transposed VHt layout [B,H,64,S]  (row=(s*B+b), col=h*64+d)
// mode 2: store f32 row-major [M,N] (final output)
struct TileSet { v16bf a0, a1, b[4]; };

static __device__ __forceinline__ void load_set(TileSet& s, const bf16* Ap, const bf16* Wp,
                                                int kk, int lane) {
  s.a0 = load_a_tile(Ap + kk, E_, lane);
  s.a1 = load_a_tile(Ap + 16 * E_ + kk, E_, lane);
#pragma unroll
  for (int t = 0; t < 4; ++t)
    s.b[t] = load_b_tile(Wp + (size_t)t * 16 * E_ + kk, E_, lane);
}

static __device__ __forceinline__ void mma_set(v8f acc[2][4], const TileSet& s) {
#pragma unroll
  for (int t = 0; t < 4; ++t) {
    acc[0][t] = wmma_bf16(s.a0, s.b[t], acc[0][t]);
    acc[1][t] = wmma_bf16(s.a1, s.b[t], acc[1][t]);
  }
}

__global__ __launch_bounds__(256) void gemm_bf16_k(const bf16* __restrict__ A,
                                                   const bf16* __restrict__ W,
                                                   const float* __restrict__ bias,
                                                   int N, int mode, int headDim,
                                                   bf16* __restrict__ outb,
                                                   float* __restrict__ outf) {
  const int lane   = threadIdx.x & 31;
  const int wave   = (blockIdx.x * blockDim.x + threadIdx.x) >> 5;
  const int ntiles = N >> 6;
  const int mt = wave / ntiles;
  const int nt = wave - mt * ntiles;
  if (mt >= (MROWS >> 5)) return;
  const int m0 = mt << 5, n0 = nt << 6;

  const bf16* Ap = A + (size_t)m0 * E_;
  const bf16* Wp = W + (size_t)n0 * E_;
  v8f acc[2][4] = {};

  TileSet sA, sB;
  load_set(sA, Ap, Wp, 0, lane);
#pragma unroll 1
  for (int kk = 0; kk < E_ - 64; kk += 64) {
    load_set(sB, Ap, Wp, kk + 32, lane);   // phase B loads in flight
    mma_set(acc, sA);                      // compute on phase A
    load_set(sA, Ap, Wp, kk + 64, lane);   // phase A reloads in flight
    mma_set(acc, sB);                      // compute on phase B
  }
  load_set(sB, Ap, Wp, E_ - 32, lane);     // tail (kk = E_-64)
  mma_set(acc, sA);
  mma_set(acc, sB);

  const int half = lane >> 4, cl = lane & 15;
#pragma unroll
  for (int mh = 0; mh < 2; ++mh) {
#pragma unroll
    for (int t = 0; t < 4; ++t) {
      const int col = n0 + t * 16 + cl;
      const float bc = bias ? bias[col] : 0.f;
#pragma unroll
      for (int j = 0; j < 8; ++j) {
        const float v  = acc[mh][t][j] + bc;
        const int row  = m0 + mh * 16 + j + half * 8;
        if (mode == 0) {
          const int tt = row >> 1, bb = row & 1;     // B_ == 2
          const int h = col / headDim, k = col - h * headDim;
          outb[(((bb * H_ + h) * T_) + tt) * headDim + k] = (bf16)v;
        } else if (mode == 1) {
          const int ss = row >> 1, bb = row & 1;
          const int h = col >> 6, d = col & 63;
          outb[(((bb * H_ + h) * HD_) + d) * S_ + ss] = (bf16)v;
        } else {
          outf[row * N + col] = v;
        }
      }
    }
  }
}

// ---------------- flash attention (per (b,h), wave = 16 query rows) ----------------
// QH[B,H,T,32] (scale*log2e pre-folded into Q weights), KH[B,H,S,32], VHt[B,H,64,S].
// Ping-pong K tiles (s-loop body holds both phases; unroll 1): next block's K loads
// are in flight across the whole softmax of the current block; V tiles issue right
// after the score WMMAs. Output O2d[(t*B+b)*E + h*64 + d] in bf16.
__global__ __launch_bounds__(256) void mla_attention_k(const bf16* __restrict__ QH,
                                                       const bf16* __restrict__ KH,
                                                       const bf16* __restrict__ VT,
                                                       bf16* __restrict__ O) {
  __shared__ bf16 plds[8][16 * 32];            // 1 KB per wave P-staging tile
  const int lane = threadIdx.x & 31;
  const int w    = threadIdx.x >> 5;
  const int tblocks = T_ / 128;                // 16
  const int bh = blockIdx.x / tblocks;
  const int tb = blockIdx.x - bh * tblocks;
  const int b = bh / H_, h = bh - b * H_;
  const int q0 = tb * 128 + w * 16;

  const bf16* Qp = QH + (size_t)(b * H_ + h) * T_ * DK_;
  const bf16* Kp = KH + (size_t)(b * H_ + h) * S_ * DK_;
  const bf16* Vp = VT + (size_t)(b * H_ + h) * HD_ * S_;

  const v16bf qa = load_a_tile(Qp + q0 * DK_, DK_, lane);  // Q tile resident all loop

  v8f acc[4] = {};
  float mrun[8], lrun[8];
#pragma unroll
  for (int j = 0; j < 8; ++j) { mrun[j] = -1e30f; lrun[j] = 0.f; }

  const int half = lane >> 4, cl = lane & 15;
  bf16* myp = &plds[w][0];

  // one 32-key block: scores -> online softmax -> P@V
  auto process = [&](const v16bf& k0, const v16bf& k1, int s0) {
    v8f z = {};
    const v8f sc0 = wmma_bf16(qa, k0, z);
    const v8f sc1 = wmma_bf16(qa, k1, z);

    const v16bf vb0 = load_b_tile(Vp + (size_t) 0 * S_ + s0, S_, lane);
    const v16bf vb1 = load_b_tile(Vp + (size_t)16 * S_ + s0, S_, lane);
    const v16bf vb2 = load_b_tile(Vp + (size_t)32 * S_ + s0, S_, lane);
    const v16bf vb3 = load_b_tile(Vp + (size_t)48 * S_ + s0, S_, lane);

#pragma unroll
    for (int j = 0; j < 8; ++j) {            // C-layout: VGPR j = row j + 8*half
      const float s0v = sc0[j], s1v = sc1[j];
      float mx = fmaxf(s0v, s1v);
#pragma unroll
      for (int d = 8; d >= 1; d >>= 1) mx = fmaxf(mx, __shfl_xor(mx, d, 16));
      const float mnew = fmaxf(mrun[j], mx);
      const float corr = __builtin_amdgcn_exp2f(mrun[j] - mnew);
      const float p0 = __builtin_amdgcn_exp2f(s0v - mnew);
      const float p1 = __builtin_amdgcn_exp2f(s1v - mnew);
      float rs = p0 + p1;
#pragma unroll
      for (int d = 8; d >= 1; d >>= 1) rs += __shfl_xor(rs, d, 16);
      lrun[j] = lrun[j] * corr + rs;
      mrun[j] = mnew;
#pragma unroll
      for (int t = 0; t < 4; ++t) acc[t][j] = acc[t][j] * corr;
      const int prow = j + half * 8;         // C-layout -> A-layout via LDS
      myp[prow * 32 + cl]      = (bf16)p0;
      myp[prow * 32 + 16 + cl] = (bf16)p1;
    }

    // same-wave LDS ordering (DS ops in-order per wave); reload P as A operand
    const v16bf pa = load_a_tile(myp, 32, lane);
    acc[0] = wmma_bf16(pa, vb0, acc[0]);
    acc[1] = wmma_bf16(pa, vb1, acc[1]);
    acc[2] = wmma_bf16(pa, vb2, acc[2]);
    acc[3] = wmma_bf16(pa, vb3, acc[3]);
  };

  // ping-pong K tiles (S_/32 = 64 blocks, even); body holds both phases
  v16bf kA0 = load_b_tile(Kp, DK_, lane);
  v16bf kA1 = load_b_tile(Kp + 16 * DK_, DK_, lane);
  v16bf kB0, kB1;
#pragma unroll 1
  for (int s0 = 0; s0 < S_ - 64; s0 += 64) {
    kB0 = load_b_tile(Kp + (s0 + 32) * DK_, DK_, lane);
    kB1 = load_b_tile(Kp + (s0 + 48) * DK_, DK_, lane);
    process(kA0, kA1, s0);
    kA0 = load_b_tile(Kp + (s0 + 64) * DK_, DK_, lane);
    kA1 = load_b_tile(Kp + (s0 + 80) * DK_, DK_, lane);
    process(kB0, kB1, s0 + 32);
  }
  kB0 = load_b_tile(Kp + (S_ - 32) * DK_, DK_, lane);
  kB1 = load_b_tile(Kp + (S_ - 16) * DK_, DK_, lane);
  process(kA0, kA1, S_ - 64);
  process(kB0, kB1, S_ - 32);

  // epilogue: normalize, emit bf16 row (t*B+b), col h*64+d
#pragma unroll
  for (int j = 0; j < 8; ++j) {
    const float inv = 1.0f / lrun[j];
    const int row = q0 + j + half * 8;
#pragma unroll
    for (int t = 0; t < 4; ++t) {
      const int col = h * HD_ + t * 16 + cl;
      O[(size_t)(row * B_ + b) * E_ + col] = (bf16)(acc[t][j] * inv);
    }
  }
}

// ---------------- host orchestration ----------------
extern "C" void kernel_launch(void* const* d_in, const int* in_sizes, int n_in,
                              void* d_out, int out_size, void* d_ws, size_t ws_size,
                              hipStream_t stream) {
  (void)in_sizes; (void)n_in; (void)out_size; (void)ws_size;
  const float* query = (const float*)d_in[0];
  const float* key   = (const float*)d_in[1];
  const float* value = (const float*)d_in[2];
  const float* Wq    = (const float*)d_in[3];
  const float* bq    = (const float*)d_in[4];
  const float* Wql   = (const float*)d_in[5];
  const float* bql   = (const float*)d_in[6];
  const float* Wkl   = (const float*)d_in[7];
  const float* bkl   = (const float*)d_in[8];
  const float* Wvl   = (const float*)d_in[9];
  const float* bvl   = (const float*)d_in[10];
  const float* Wqh   = (const float*)d_in[11];
  const float* bqh   = (const float*)d_in[12];
  const float* Wkh   = (const float*)d_in[13];
  const float* bkh   = (const float*)d_in[14];
  const float* Wvh   = (const float*)d_in[15];
  const float* bvh   = (const float*)d_in[16];
  const float* Wo    = (const float*)d_in[17];
  const float* bo    = (const float*)d_in[18];
  float* out = (float*)d_out;

  char* ws = (char*)d_ws;
  size_t off = 0;
  auto alloc = [&](size_t bytes) -> void* {
    void* p = ws + off;
    off = (off + bytes + 255) & ~(size_t)255;
    return p;
  };

  bf16*  qbf    = (bf16*)alloc((size_t)MROWS * E_ * 2);
  bf16*  kbf    = (bf16*)alloc((size_t)MROWS * E_ * 2);
  bf16*  vbf    = (bf16*)alloc((size_t)MROWS * E_ * 2);
  bf16*  wobf   = (bf16*)alloc((size_t)E_ * E_ * 2);
  float* tmpQ   = (float*)alloc((size_t)DCP_ * E_ * 4);
  float* bq_eff = (float*)alloc((size_t)DCP_ * 4);
  bf16*  Wqc    = (bf16*)alloc((size_t)H_ * DK_ * E_ * 2);
  bf16*  Wkc    = (bf16*)alloc((size_t)H_ * DK_ * E_ * 2);
  bf16*  Wvc    = (bf16*)alloc((size_t)H_ * HD_ * E_ * 2);
  float* bqc    = (float*)alloc((size_t)H_ * DK_ * 4);
  float* bkc    = (float*)alloc((size_t)H_ * DK_ * 4);
  float* bvc    = (float*)alloc((size_t)H_ * HD_ * 4);
  bf16*  QHb    = (bf16*)alloc((size_t)B_ * H_ * T_ * DK_ * 2);
  bf16*  KHb    = (bf16*)alloc((size_t)B_ * H_ * S_ * DK_ * 2);
  bf16*  VTb    = (bf16*)alloc((size_t)B_ * H_ * HD_ * S_ * 2);
  bf16*  O2d    = (bf16*)alloc((size_t)MROWS * E_ * 2);

  const float QSCALE = 0.17677669529663687f /* 1/sqrt(32) */ * 1.4426950408889634f /* log2e */;

  // activations + output weight -> bf16
  f32_to_bf16_k<<<1024, 256, 0, stream>>>(query, qbf, MROWS * E_);
  f32_to_bf16_k<<<1024, 256, 0, stream>>>(key,   kbf, MROWS * E_);
  f32_to_bf16_k<<<1024, 256, 0, stream>>>(value, vbf, MROWS * E_);
  f32_to_bf16_k<<<1024, 256, 0, stream>>>(Wo,    wobf, E_ * E_);

  // fold weight chains: Wqc = QSCALE * Wqh∘Wql∘Wq ; Wkc = Wkh∘Wkl ; Wvc = Wvh∘Wvl
  fold_bias_k<<<1, 128, 0, stream>>>(Wql, bq, bql, DCP_, E_, 1.0f, bq_eff);
  combine_w_k<<<dim3(DCP_,    E_ / 256), 256, 0, stream>>>(Wql, Wq,   E_,  E_, 1.0f,   tmpQ, nullptr);
  combine_w_k<<<dim3(H_*DK_,  E_ / 256), 256, 0, stream>>>(Wqh, tmpQ, DCP_, E_, QSCALE, nullptr, Wqc);
  combine_w_k<<<dim3(H_*DK_,  E_ / 256), 256, 0, stream>>>(Wkh, Wkl,  DC_,  E_, 1.0f,   nullptr, Wkc);
  combine_w_k<<<dim3(H_*HD_,  E_ / 256), 256, 0, stream>>>(Wvh, Wvl,  DC_,  E_, 1.0f,   nullptr, Wvc);
  fold_bias_k<<<2, 256, 0, stream>>>(Wqh, bq_eff, bqh, H_ * DK_, DCP_, QSCALE, bqc);
  fold_bias_k<<<2, 256, 0, stream>>>(Wkh, bkl,    bkh, H_ * DK_, DC_,  1.0f,   bkc);
  fold_bias_k<<<4, 256, 0, stream>>>(Wvh, bvl,    bvh, H_ * HD_, DC_,  1.0f,   bvc);

  // fused projections (WMMA, 32x64 wave tiles): QH/KH [B,H,rows,32], VHt [B,H,64,S]
  gemm_bf16_k<<<128, 256, 0, stream>>>(qbf, Wqc, bqc, H_ * DK_, 0, DK_, QHb, nullptr);
  gemm_bf16_k<<<128, 256, 0, stream>>>(kbf, Wkc, bkc, H_ * DK_, 0, DK_, KHb, nullptr);
  gemm_bf16_k<<<256, 256, 0, stream>>>(vbf, Wvc, bvc, H_ * HD_, 1, HD_, VTb, nullptr);

  // flash attention (WMMA scores + WMMA P·V)
  mla_attention_k<<<B_ * H_ * (T_ / 128), 256, 0, stream>>>(QHb, KHb, VTb, O2d);

  // output projection (WMMA) -> f32 [T,B,E]
  gemm_bf16_k<<<256, 256, 0, stream>>>(O2d, wobf, bo, E_, 2, 0, nullptr, out);
}
